// GraphSAGE_1898375544834
// MI455X (gfx1250) — matrix operand order repaired and checked by reading, not compiled, not run on hardware
//
#include <hip/hip_runtime.h>
#include <math.h>

#define D 128          // feature width (in & hidden)
#define DOUT 40        // output classes
#define DOUT_PAD 48    // padded to 3 x 16 WMMA tiles

typedef __attribute__((ext_vector_type(16))) _Float16 v16h;
typedef __attribute__((ext_vector_type(8)))  _Float16 v8h;
typedef __attribute__((ext_vector_type(8)))  float    v8f;

// ---------------------------------------------------------------------------
// WMMA fragment loaders (wave32, 16x16x32 f16 -> f32)
// A 16x32: lane holds row (lane&15); K elements [kb..kb+7, 16+kb..16+kb+7],
//          kb = (lane>>4)*8  (per ISA 7.12.2 "16-bit A-Matrix 16x32")
// B 32x16: lane holds weight-row/out-col (lane&15); K elements [kb..kb+7, 16+kb..16+kb+7],
//          kb = (lane>>4)*8, same K order as A
// ---------------------------------------------------------------------------
__device__ __forceinline__ v16h frag_a(const _Float16* p) {
  v8h lo = *(const v8h*)p;           // K = kb .. kb+7
  v8h hi = *(const v8h*)(p + 16);    // K = 16+kb .. 16+kb+7
  v16h f;
#pragma unroll
  for (int i = 0; i < 8; ++i) { f[i] = lo[i]; f[i + 8] = hi[i]; }
  return f;
}
__device__ __forceinline__ v16h frag_b(const _Float16* p) {
  v8h lo = *(const v8h*)p;           // K = kb .. kb+7
  v8h hi = *(const v8h*)(p + 16);    // K = 16+kb .. 16+kb+7
  v16h f;
#pragma unroll
  for (int i = 0; i < 8; ++i) { f[i] = lo[i]; f[i + 8] = hi[i]; }
  return f;
}

// Fused SAGE GEMM: out[m, n] = sum_k xself[m,k]*Wself[n,k] + xneigh[m,k]*Wneigh[n,k]
// One wave = one 16x16 output tile; blockDim.x = 32 * n_tiles; grid.x = N/16.
__global__ void sage_gemm_wmma(const _Float16* __restrict__ xself,
                               const _Float16* __restrict__ xneigh,
                               const _Float16* __restrict__ Wself,   // [padrows x 128] f16
                               const _Float16* __restrict__ Wneigh,  // [padrows x 128] f16
                               float* __restrict__ out, int dout) {
  const int lane = threadIdx.x & 31;
  const int tile = threadIdx.x >> 5;                  // output column tile
  const int mrow = (blockIdx.x << 4) + (lane & 15);   // A row for this lane
  const int wrow = (tile << 4) + (lane & 15);         // weight row = output col
  const int kb   = (lane >> 4) << 3;                  // A frag K base (0 or 8)
  const int kh   = (lane >> 4) << 3;                  // B frag K base (0 or 8)

  const _Float16* aS = xself  + (size_t)mrow * D + kb;
  const _Float16* aN = xneigh + (size_t)mrow * D + kb;
  const _Float16* bS = Wself  + (size_t)wrow * D + kh;
  const _Float16* bN = Wneigh + (size_t)wrow * D + kh;

  v8f acc = {};
#pragma unroll
  for (int k0 = 0; k0 < D; k0 += 32) {
    v16h a0 = frag_a(aS + k0);
    v16h b0 = frag_b(bS + k0);
    acc = __builtin_amdgcn_wmma_f32_16x16x32_f16(false, a0, false, b0,
                                                 (short)0, acc, false, false);
    v16h a1 = frag_a(aN + k0);
    v16h b1 = frag_b(bN + k0);
    acc = __builtin_amdgcn_wmma_f32_16x16x32_f16(false, a1, false, b1,
                                                 (short)0, acc, false, false);
  }

  // C/D layout: element r -> M = r + 8*(lane>=16), N = lane&15
  const int col = (tile << 4) + (lane & 15);
  if (col < dout) {
    const int mbase = (blockIdx.x << 4) + ((lane >> 4) << 3);
#pragma unroll
    for (int r = 0; r < 8; ++r)
      out[(size_t)(mbase + r) * dout + col] = acc[r];
  }
}

// ---------------------------------------------------------------------------
// Irregular graph kernels
// ---------------------------------------------------------------------------
__global__ void zero_f32(float* __restrict__ p, long n) {
  long i = (long)blockIdx.x * blockDim.x + threadIdx.x;
  if (i < n) p[i] = 0.0f;
}

__global__ void deg_count(const int* __restrict__ dst, float* __restrict__ deg, int e) {
  int i = blockIdx.x * blockDim.x + threadIdx.x;
  if (i < e) unsafeAtomicAdd(&deg[dst[i]], 1.0f);
}

__global__ void make_invdeg(const float* __restrict__ deg, float* __restrict__ inv, int n) {
  int i = blockIdx.x * blockDim.x + threadIdx.x;
  if (i < n) inv[i] = 1.0f / fmaxf(deg[i], 1.0f);
}

// One wave per edge; lane moves float4 (128 floats / 32 lanes) with f32 atomics.
__global__ void spmm_scatter(const float* __restrict__ x, const int* __restrict__ src,
                             const int* __restrict__ dst, float* __restrict__ agg, int e) {
  int edge = blockIdx.x * (blockDim.x >> 5) + (threadIdx.x >> 5);
  if (edge >= e) return;
  int lane = threadIdx.x & 31;
  int s = __ldg(&src[edge]);
  int d = __ldg(&dst[edge]);
  const float4 v = *(const float4*)(x + (size_t)s * D + lane * 4);
  float* o = agg + (size_t)d * D + lane * 4;
  unsafeAtomicAdd(o + 0, v.x);
  unsafeAtomicAdd(o + 1, v.y);
  unsafeAtomicAdd(o + 2, v.z);
  unsafeAtomicAdd(o + 3, v.w);
}

// agg_f16 = (agg * invdeg[row]) cast to f16
__global__ void scale_cast(const float* __restrict__ agg, const float* __restrict__ inv,
                           _Float16* __restrict__ out, long n) {
  long i = (long)blockIdx.x * blockDim.x + threadIdx.x;
  if (i < n) out[i] = (_Float16)(agg[i] * inv[i >> 7]);
}

__global__ void cast_f16(const float* __restrict__ in, _Float16* __restrict__ out, long n) {
  long i = (long)blockIdx.x * blockDim.x + threadIdx.x;
  if (i < n) out[i] = (_Float16)in[i];
}

// f32 [rows x 128] -> f16 [padrows x 128], zero-padding extra rows
__global__ void cast_weight(const float* __restrict__ w, _Float16* __restrict__ wh,
                            int rows, int padrows) {
  int i = blockIdx.x * blockDim.x + threadIdx.x;
  if (i >= padrows * D) return;
  int r = i >> 7;
  wh[i] = (r < rows) ? (_Float16)w[i] : (_Float16)0.0f;
}

// ---------------------------------------------------------------------------
// BatchNorm (over N rows, 128 cols): stats then apply + ReLU (+ f16 copy)
// ---------------------------------------------------------------------------
__global__ void bn_stats(const float* __restrict__ h, float* __restrict__ sums, int nrows) {
  int col = threadIdx.x & (D - 1);
  int sub = threadIdx.x >> 7;  // 0..1
  float s = 0.0f, q = 0.0f;
  for (int r = blockIdx.x * 2 + sub; r < nrows; r += gridDim.x * 2) {
    float v = h[(size_t)r * D + col];
    s += v; q += v * v;
  }
  __shared__ float ls[256], lq[256];
  ls[threadIdx.x] = s; lq[threadIdx.x] = q;
  __syncthreads();
  if (threadIdx.x < D) {
    unsafeAtomicAdd(&sums[threadIdx.x],     ls[threadIdx.x] + ls[threadIdx.x + D]);
    unsafeAtomicAdd(&sums[D + threadIdx.x], lq[threadIdx.x] + lq[threadIdx.x + D]);
  }
}

__global__ void bn_apply(float* __restrict__ h, _Float16* __restrict__ hf16,
                         const float* __restrict__ sums,
                         const float* __restrict__ gamma, const float* __restrict__ beta,
                         long n, float rn) {
  long i = (long)blockIdx.x * blockDim.x + threadIdx.x;
  if (i >= n) return;
  int col = (int)(i & (D - 1));
  float mu  = sums[col] * rn;
  float var = sums[D + col] * rn - mu * mu;
  float v = (h[i] - mu) * rsqrtf(var + 1e-5f) * gamma[col] + beta[col];
  v = fmaxf(v, 0.0f);
  h[i] = v;
  hf16[i] = (_Float16)v;
}

// ---------------------------------------------------------------------------
// log_softmax over 40 classes; one wave per row; in-place safe.
// ---------------------------------------------------------------------------
__global__ void log_softmax40(const float* __restrict__ in, float* __restrict__ out, int nrows) {
  int row = blockIdx.x * (blockDim.x >> 5) + (threadIdx.x >> 5);
  if (row >= nrows) return;
  int lane = threadIdx.x & 31;
  const float* p = in + (size_t)row * DOUT;
  float v0 = (lane < DOUT)     ? p[lane]      : -INFINITY;   // cols 0..31
  float v1 = (lane < DOUT - 32)? p[32 + lane] : -INFINITY;   // cols 32..39
  float m = fmaxf(v0, v1);
#pragma unroll
  for (int off = 16; off; off >>= 1) m = fmaxf(m, __shfl_xor(m, off, 32));
  float s = expf(v0 - m) + ((lane < DOUT - 32) ? expf(v1 - m) : 0.0f);
#pragma unroll
  for (int off = 16; off; off >>= 1) s += __shfl_xor(s, off, 32);
  float ls = logf(s);
  float* q = out + (size_t)row * DOUT;
  q[lane] = v0 - m - ls;
  if (lane < DOUT - 32) q[32 + lane] = v1 - m - ls;
}

// ---------------------------------------------------------------------------
extern "C" void kernel_launch(void* const* d_in, const int* in_sizes, int n_in,
                              void* d_out, int out_size, void* d_ws, size_t ws_size,
                              hipStream_t stream) {
  const float* feat = (const float*)d_in[0];
  const float* Ws1  = (const float*)d_in[1];
  const float* Wn1  = (const float*)d_in[2];
  const float* g1   = (const float*)d_in[3];
  const float* b1   = (const float*)d_in[4];
  const float* Ws2  = (const float*)d_in[5];
  const float* Wn2  = (const float*)d_in[6];
  const float* g2   = (const float*)d_in[7];
  const float* b2   = (const float*)d_in[8];
  const float* Ws3  = (const float*)d_in[9];
  const float* Wn3  = (const float*)d_in[10];
  const int*   src  = (const int*)d_in[11];
  const int*   dst  = (const int*)d_in[12];

  const int E_ = in_sizes[11];
  const int N_ = in_sizes[0] / D;          // 100000 (divisible by 16)
  const long ND = (long)N_ * D;

  // --- workspace carve-out (256B aligned) ---
  char* cur = (char*)d_ws;
  auto alloc = [&](size_t bytes) -> char* {
    char* p = cur;
    cur += (bytes + 255) & ~(size_t)255;
    return p;
  };
  float*    deg   = (float*)   alloc((size_t)N_ * 4);
  float*    inv   = (float*)   alloc((size_t)N_ * 4);
  float*    agg   = (float*)   alloc((size_t)ND * 4);
  _Float16* aggh  = (_Float16*)alloc((size_t)ND * 2);
  _Float16* xh    = (_Float16*)alloc((size_t)ND * 2);
  float*    h     = (float*)   alloc((size_t)ND * 4);
  _Float16* ws1h  = (_Float16*)alloc((size_t)D * D * 2);
  _Float16* wn1h  = (_Float16*)alloc((size_t)D * D * 2);
  _Float16* ws2h  = (_Float16*)alloc((size_t)D * D * 2);
  _Float16* wn2h  = (_Float16*)alloc((size_t)D * D * 2);
  _Float16* ws3h  = (_Float16*)alloc((size_t)DOUT_PAD * D * 2);
  _Float16* wn3h  = (_Float16*)alloc((size_t)DOUT_PAD * D * 2);
  float*    bns   = (float*)   alloc(2 * D * 4);

  const int TB = 256;
  auto g1d = [&](long n) { return (unsigned)((n + TB - 1) / TB); };
  const unsigned gSp = (unsigned)((E_ + 7) / 8);        // spmm: 8 waves/block
  const unsigned gMM = (unsigned)(N_ / 16);             // gemm: 16 rows/block
  const float rn = 1.0f / (float)N_;

  // --- degrees ---
  zero_f32<<<g1d(N_), TB, 0, stream>>>(deg, N_);
  deg_count<<<g1d(E_), TB, 0, stream>>>(dst, deg, E_);
  make_invdeg<<<g1d(N_), TB, 0, stream>>>(deg, inv, N_);

  // --- weight casts (f32 -> f16, layer3 zero-padded to 48 rows) ---
  cast_weight<<<g1d((long)D * D), TB, 0, stream>>>(Ws1, ws1h, D, D);
  cast_weight<<<g1d((long)D * D), TB, 0, stream>>>(Wn1, wn1h, D, D);
  cast_weight<<<g1d((long)D * D), TB, 0, stream>>>(Ws2, ws2h, D, D);
  cast_weight<<<g1d((long)D * D), TB, 0, stream>>>(Wn2, wn2h, D, D);
  cast_weight<<<g1d((long)DOUT_PAD * D), TB, 0, stream>>>(Ws3, ws3h, DOUT, DOUT_PAD);
  cast_weight<<<g1d((long)DOUT_PAD * D), TB, 0, stream>>>(Wn3, wn3h, DOUT, DOUT_PAD);

  // --- layer 1 ---
  cast_f16<<<g1d(ND), TB, 0, stream>>>(feat, xh, ND);
  zero_f32<<<g1d(ND), TB, 0, stream>>>(agg, ND);
  spmm_scatter<<<gSp, TB, 0, stream>>>(feat, src, dst, agg, E_);
  scale_cast<<<g1d(ND), TB, 0, stream>>>(agg, inv, aggh, ND);
  sage_gemm_wmma<<<gMM, 8 * 32, 0, stream>>>(xh, aggh, ws1h, wn1h, h, D);
  zero_f32<<<1, TB, 0, stream>>>(bns, 2 * D);
  bn_stats<<<1024, TB, 0, stream>>>(h, bns, N_);
  bn_apply<<<g1d(ND), TB, 0, stream>>>(h, xh, bns, g1, b1, ND, rn);

  // --- layer 2 ---
  zero_f32<<<g1d(ND), TB, 0, stream>>>(agg, ND);
  spmm_scatter<<<gSp, TB, 0, stream>>>(h, src, dst, agg, E_);
  scale_cast<<<g1d(ND), TB, 0, stream>>>(agg, inv, aggh, ND);
  sage_gemm_wmma<<<gMM, 8 * 32, 0, stream>>>(xh, aggh, ws2h, wn2h, h, D);
  zero_f32<<<1, TB, 0, stream>>>(bns, 2 * D);
  bn_stats<<<1024, TB, 0, stream>>>(h, bns, N_);
  bn_apply<<<g1d(ND), TB, 0, stream>>>(h, xh, bns, g2, b2, ND, rn);

  // --- layer 3 (straight into d_out, then in-place log_softmax) ---
  zero_f32<<<g1d(ND), TB, 0, stream>>>(agg, ND);
  spmm_scatter<<<gSp, TB, 0, stream>>>(h, src, dst, agg, E_);
  scale_cast<<<g1d(ND), TB, 0, stream>>>(agg, inv, aggh, ND);
  sage_gemm_wmma<<<gMM, 3 * 32, 0, stream>>>(xh, aggh, ws3h, wn3h, (float*)d_out, DOUT);
  log_softmax40<<<(unsigned)((N_ + 7) / 8), TB, 0, stream>>>((const float*)d_out,
                                                             (float*)d_out, N_);
}
